// RCNNRoIhead_60000693125653
// MI455X (gfx1250) — matrix operand order repaired and checked by reading, not compiled, not run on hardware
//
#include <hip/hip_runtime.h>
#include <math.h>

// ---------------- problem constants (from reference) ----------------
#define BATCH   2
#define CCH     512
#define HH      50
#define WW      50
#define NROIS   512
#define PP      7
#define RSCALE  (1.0f/16.0f)
#define FLAT    (CCH*PP*PP)      // 25088
#define MID     4096
#define NCLS    21
#define NLOC    (NCLS*4)         // 84
#define NHEAD   (NCLS + NLOC)    // 105

typedef __bf16 bf16;
typedef __attribute__((ext_vector_type(8)))  __bf16 v8bf;
typedef __attribute__((ext_vector_type(16))) __bf16 v16bf;
typedef __attribute__((ext_vector_type(8)))  float  v8f;

// exact parameter type of the async LDS-DMA builtin (per hipcc diagnostic)
typedef int vint4 __attribute__((__vector_size__(4 * sizeof(int))));

#define GLOBAL_AS __attribute__((address_space(1)))
#define LDS_AS    __attribute__((address_space(3)))

#if __has_builtin(__builtin_amdgcn_global_load_async_to_lds_b128)
#define HAVE_ASYNC_LDS 1
#else
#define HAVE_ASYNC_LDS 0
#endif

static __device__ __forceinline__ v16bf cat8(v8bf lo, v8bf hi) {
  return __builtin_shufflevector(lo, hi, 0,1,2,3,4,5,6,7,8,9,10,11,12,13,14,15);
}
static __device__ __forceinline__ float f4get(const float4& v, int i) {
  const float* p = &v.x;
  return p[i];
}

// ---------------- 1) RoIPool (torchvision semantics) -> bf16 ----------------
__global__ __launch_bounds__(256)
void roipool_kernel(const float* __restrict__ feat, const float* __restrict__ rois,
                    const int* __restrict__ ridx, bf16* __restrict__ pooled)
{
  const int roi = blockIdx.x;
  const float4 r = *(const float4*)(rois + 4*roi);
  const int b = ridx[roi];
  const int sw = (int)floorf(r.x * RSCALE + 0.5f);
  const int sh = (int)floorf(r.y * RSCALE + 0.5f);
  const int ew = (int)floorf(r.z * RSCALE + 0.5f);
  const int eh = (int)floorf(r.w * RSCALE + 0.5f);
  const float bin_h = (float)max(eh - sh + 1, 1) / (float)PP;
  const float bin_w = (float)max(ew - sw + 1, 1) / (float)PP;
  const float* fb = feat + (size_t)b * CCH * HH * WW;

  for (int idx = threadIdx.x; idx < CCH * PP * PP; idx += blockDim.x) {
    const int pw = idx % PP;
    const int ph = (idx / PP) % PP;
    const int c  = idx / (PP * PP);
    int hs = min(max((int)floorf((float)ph * bin_h)     + sh, 0), HH);
    int he = min(max((int)ceilf ((float)(ph+1) * bin_h) + sh, 0), HH);
    int ws = min(max((int)floorf((float)pw * bin_w)     + sw, 0), WW);
    int we = min(max((int)ceilf ((float)(pw+1) * bin_w) + sw, 0), WW);
    float m = -INFINITY;
    const float* fc = fb + (size_t)c * HH * WW;
    for (int ih = hs; ih < he; ++ih)
      for (int iw = ws; iw < we; ++iw)
        m = fmaxf(m, fc[ih * WW + iw]);
    if (he <= hs || we <= ws) m = 0.0f;   // empty bin -> 0
    pooled[(size_t)roi * FLAT + idx] = (bf16)m;
  }
}

// ---------------- 2) bf16 WMMA GEMM: Out = relu(A @ Bw + bias) --------------
// A: M x K bf16 row-major.  Bw: K x N fp32 row-major (converted to bf16 on the
// fly).  Out: M x N bf16.  Block tile 128x128, K-step 32, 256 thr = 8 waves,
// wave tile 64x32.  Double-buffered LDS; A staged via async LDS-DMA.
template<bool RELU>
__global__ __launch_bounds__(256)
void gemm_bf16_wmma(const bf16* __restrict__ A, const float* __restrict__ Bw,
                    const float* __restrict__ bias, bf16* __restrict__ Out,
                    int M, int N, int K)
{
  __shared__ bf16 As[2][128][40];   // row-major M x K tile (80B row stride)
  __shared__ bf16 Bs[2][128][40];   // n-major  N x K tile (transposed on store)

  const int tid   = threadIdx.x;
  const int lane  = tid & 31;
  const int wave  = tid >> 5;
  const int lrow  = lane & 15;
  const int lhalf = lane >> 4;
  const int m0 = blockIdx.y * 128;
  const int n0 = blockIdx.x * 128;
  const int m_off = (wave >> 2) * 64; // 2 waves along M
  const int n_off = (wave & 3) * 32;  // 4 waves along N

  // A staging: thread covers 2 x 16B chunks of the 128x32 bf16 tile
  const int ar0 = (tid * 2)     >> 2, ac0 = ((tid * 2)     & 3) * 8;
  const int ar1 = (tid * 2 + 1) >> 2, ac1 = ((tid * 2 + 1) & 3) * 8;
  // B staging: thread covers a 4(k) x 4(n) fp32 micro-tile
  const int bnc = (tid & 31) * 4;     // n offset 0..124
  const int bkc = (tid >> 5) * 4;     // k offset 0..28

  v8f acc[4][2];
  #pragma unroll
  for (int i = 0; i < 4; ++i)
    #pragma unroll
    for (int j = 0; j < 2; ++j)
      acc[i][j] = v8f{};

  float4 breg[4];

  auto stageA = [&](int k0, int buf) {
#if HAVE_ASYNC_LDS
    __builtin_amdgcn_global_load_async_to_lds_b128(
        (GLOBAL_AS vint4*)(A + (size_t)(m0 + ar0) * K + k0 + ac0),
        (LDS_AS vint4*)&As[buf][ar0][ac0], 0, 0);
    __builtin_amdgcn_global_load_async_to_lds_b128(
        (GLOBAL_AS vint4*)(A + (size_t)(m0 + ar1) * K + k0 + ac1),
        (LDS_AS vint4*)&As[buf][ar1][ac1], 0, 0);
#else
    *(uint4*)&As[buf][ar0][ac0] = *(const uint4*)(A + (size_t)(m0 + ar0) * K + k0 + ac0);
    *(uint4*)&As[buf][ar1][ac1] = *(const uint4*)(A + (size_t)(m0 + ar1) * K + k0 + ac1);
#endif
  };
  auto waitA = [&]() {
#if HAVE_ASYNC_LDS
    asm volatile("s_wait_asynccnt 0x0" ::: "memory");
#endif
  };
  auto loadB = [&](int k0) {
    #pragma unroll
    for (int j = 0; j < 4; ++j)
      breg[j] = *(const float4*)(Bw + (size_t)(k0 + bkc + j) * N + n0 + bnc);
  };
  auto storeB = [&](int buf) {
    #pragma unroll
    for (int i = 0; i < 4; ++i) {     // n within micro-tile
      bf16 p[4] = { (bf16)f4get(breg[0], i), (bf16)f4get(breg[1], i),
                    (bf16)f4get(breg[2], i), (bf16)f4get(breg[3], i) };
      *(uint2*)&Bs[buf][bnc + i][bkc] = *(uint2*)p;   // 8B: 4 consecutive K
    }
  };

  const int T = K / 32;

  // prologue: stage tile 0
  stageA(0, 0);
  loadB(0);
  storeB(0);
  waitA();
  __syncthreads();

  for (int kt = 0; kt < T; ++kt) {
    const int cur = kt & 1;
    const bool more = (kt + 1 < T);

    // issue next tile's loads first so they overlap the WMMA work below
    if (more) {
      stageA((kt + 1) * 32, 1 - cur);
      loadB((kt + 1) * 32);
      if (kt + 2 < T)   // warm L2 for the tile after next (global_prefetch_b8)
        __builtin_prefetch(Bw + (size_t)((kt + 2) * 32 + bkc) * N + n0 + bnc, 0, 0);
    }

    // ---- fragments per ISA 16-bit layouts, from buffer `cur` ----
    v16bf afrag[4];
    #pragma unroll
    for (int mi = 0; mi < 4; ++mi) {
      const bf16* rowp = &As[cur][m_off + mi * 16 + lrow][0];
      afrag[mi] = cat8(*(const v8bf*)(rowp + lhalf * 8),
                       *(const v8bf*)(rowp + 16 + lhalf * 8));
    }
    v16bf bfrag[2];
    #pragma unroll
    for (int ni = 0; ni < 2; ++ni) {
      const bf16* colp = &Bs[cur][n_off + ni * 16 + lrow][0];
      const v8bf* p = (const v8bf*)(colp + lhalf * 8);
      bfrag[ni] = cat8(p[0], p[2]);
    }

    #pragma unroll
    for (int mi = 0; mi < 4; ++mi)
      #pragma unroll
      for (int ni = 0; ni < 2; ++ni)
        acc[mi][ni] = __builtin_amdgcn_wmma_f32_16x16x32_bf16(
            false, afrag[mi], false, bfrag[ni],
            (short)0, acc[mi][ni], false, false);

    // finish staging the next buffer, then one barrier flips the pipeline
    if (more) {
      storeB(1 - cur);
      waitA();
    }
    __syncthreads();
  }

  // ---- epilogue: bias + optional ReLU, store bf16 (C/D lane layout) ----
  #pragma unroll
  for (int mi = 0; mi < 4; ++mi) {
    #pragma unroll
    for (int ni = 0; ni < 2; ++ni) {
      const int gn = n0 + n_off + ni * 16 + lrow;
      const float bia = bias[gn];
      const int gm_base = m0 + m_off + mi * 16 + lhalf * 8;
      #pragma unroll
      for (int v = 0; v < 8; ++v) {
        float val = acc[mi][ni][v] + bia;
        if (RELU) val = fmaxf(val, 0.0f);
        Out[(size_t)(gm_base + v) * N + gn] = (bf16)val;
      }
    }
  }
}

// ---------------- 3) small heads: scores(21) + locs(84) --------------------
__global__ __launch_bounds__(256)
void head_kernel(const bf16* __restrict__ X, const float* __restrict__ Wc,
                 const float* __restrict__ bc, const float* __restrict__ Wl,
                 const float* __restrict__ bl, float* __restrict__ out)
{
  const int gid = blockIdx.x * blockDim.x + threadIdx.x;
  if (gid >= NROIS * NHEAD) return;
  const int m = gid / NHEAD;
  const int c = gid % NHEAD;
  const bf16* x = X + (size_t)m * MID;

  const float* w;
  int ldw, col;
  float acc;
  if (c < NCLS) { w = Wc; ldw = NCLS; col = c;        acc = bc[col]; }
  else          { w = Wl; ldw = NLOC; col = c - NCLS; acc = bl[col]; }

  for (int k = 0; k < MID; k += 8) {
    const v8bf xa = *(const v8bf*)(x + k);   // 16B activation chunk (L2 resident)
    #pragma unroll
    for (int j = 0; j < 8; ++j)
      acc += (float)xa[j] * w[(size_t)(k + j) * ldw + col];
  }

  if (c < NCLS) out[(size_t)m * NCLS + col] = acc;
  else          out[(size_t)NROIS * NCLS + (size_t)m * NLOC + col] = acc;
}

// ---------------- launcher -------------------------------------------------
extern "C" void kernel_launch(void* const* d_in, const int* in_sizes, int n_in,
                              void* d_out, int out_size, void* d_ws, size_t ws_size,
                              hipStream_t stream)
{
  const float* h    = (const float*)d_in[0];
  const float* rois = (const float*)d_in[1];
  const int*   ridx = (const int*)  d_in[2];
  const float* W1   = (const float*)d_in[3];
  const float* b1   = (const float*)d_in[4];
  const float* W2   = (const float*)d_in[5];
  const float* b2   = (const float*)d_in[6];
  const float* Wc   = (const float*)d_in[7];
  const float* bc   = (const float*)d_in[8];
  const float* Wl   = (const float*)d_in[9];
  const float* bl   = (const float*)d_in[10];
  float* out = (float*)d_out;

  // workspace: pooled (bf16 512x25088) | x1 (bf16 512x4096) | x2 (bf16 512x4096)
  char* ws = (char*)d_ws;
  bf16* pooled = (bf16*)ws;
  bf16* x1     = (bf16*)(ws + (size_t)NROIS * FLAT * sizeof(bf16));
  bf16* x2     = (bf16*)(ws + (size_t)NROIS * FLAT * sizeof(bf16)
                            + (size_t)NROIS * MID  * sizeof(bf16));

  roipool_kernel<<<NROIS, 256, 0, stream>>>(h, rois, ridx, pooled);

  dim3 blk(256);
  dim3 grd(MID / 128, NROIS / 128);   // 32 x 4
  gemm_bf16_wmma<true><<<grd, blk, 0, stream>>>(pooled, W1, b1, x1, NROIS, MID, FLAT);
  gemm_bf16_wmma<true><<<grd, blk, 0, stream>>>(x1,     W2, b2, x2, NROIS, MID, MID);

  const int tot = NROIS * NHEAD;
  head_kernel<<<(tot + 255) / 256, 256, 0, stream>>>(x2, Wc, bc, Wl, bl, out);
}
